// MoEBlock_85504208929179
// MI455X (gfx1250) — matrix operand order repaired and checked by reading, not compiled, not run on hardware
//
#include <hip/hip_runtime.h>
#include <hip/hip_bf16.h>

// ---------------------------------------------------------------------------
// MoE block for MI455X (gfx1250, wave32).
// Route top-2, then fused FFN (w1->relu->w2) per expert tile using
// v_wmma_f32_16x16x32_bf16 with fp32 accumulation.
// MT=64 token tiles, 512-thread blocks (16 waves): every weight B-fragment
// feeds 4 WMMAs, cutting L2 weight traffic 4x vs the MT=16 tiling.
// ---------------------------------------------------------------------------

typedef __attribute__((ext_vector_type(16))) __bf16 v16bf;
typedef __attribute__((ext_vector_type(8)))  float  v8f;

#define B_    4
#define T_    1024
#define D_    1024
#define H_    4096
#define E_    8
#define NTOK  (B_ * T_)

#define MT        64     // tokens per block tile (4 WMMA M-subtiles)
#define HC        256    // H chunk held in LDS per iteration
#define BLOCK     512    // 16 waves
#define NCOL_W    64     // output (D) columns owned by each wave = 4 N-tiles

// ---------------------------------------------------------------------------
// Kernel 0: zero output + routing counters (d_out is poisoned by harness)
// ---------------------------------------------------------------------------
__global__ void moe_zero_kernel(float* __restrict__ out, int n,
                                int* __restrict__ counts) {
    int i = blockIdx.x * blockDim.x + threadIdx.x;
    if (i < n) out[i] = 0.0f;
    if (blockIdx.x == 0 && threadIdx.x < E_) counts[threadIdx.x] = 0;
}

// ---------------------------------------------------------------------------
// Kernel 1: gate = softmax(x @ gate_w + gate_b), top-2 routing lists.
// One wave (32 lanes) per token.
// ---------------------------------------------------------------------------
__global__ void moe_gate_topk(const float* __restrict__ x,
                              const float* __restrict__ gate_w,
                              const float* __restrict__ gate_b,
                              int*   __restrict__ counts,
                              int*   __restrict__ tokidx,
                              float* __restrict__ tokgate) {
    const int token = blockIdx.x * (blockDim.x >> 5) + (threadIdx.x >> 5);
    const int lane  = threadIdx.x & 31;
    if (token >= NTOK) return;

    const float* xr = x + (size_t)token * D_;
    float acc[E_];
#pragma unroll
    for (int e = 0; e < E_; ++e) acc[e] = 0.0f;

    for (int i = 0; i < D_ / 32; ++i) {
        const int d    = lane + i * 32;              // coalesced across the wave
        const float xv = xr[d];
        const float* gw = gate_w + (size_t)d * E_;   // 8 contiguous floats
#pragma unroll
        for (int e = 0; e < E_; ++e) acc[e] += xv * gw[e];
    }
    // wave32 butterfly reduction
#pragma unroll
    for (int e = 0; e < E_; ++e) {
#pragma unroll
        for (int off = 16; off > 0; off >>= 1)
            acc[e] += __shfl_xor(acc[e], off, 32);
    }

    if (lane == 0) {
        float m = -1e30f;
#pragma unroll
        for (int e = 0; e < E_; ++e) { acc[e] += gate_b[e]; m = fmaxf(m, acc[e]); }
        float p[E_], s = 0.0f;
#pragma unroll
        for (int e = 0; e < E_; ++e) { p[e] = __expf(acc[e] - m); s += p[e]; }
        const float inv = 1.0f / s;

        int i0 = 0;
#pragma unroll
        for (int e = 1; e < E_; ++e) if (p[e] > p[i0]) i0 = e;
        int i1 = (i0 == 0) ? 1 : 0;
#pragma unroll
        for (int e = 0; e < E_; ++e) if (e != i0 && p[e] > p[i1]) i1 = e;

        int s0 = atomicAdd(&counts[i0], 1);
        tokidx [i0 * NTOK + s0] = token;
        tokgate[i0 * NTOK + s0] = p[i0] * inv;
        int s1 = atomicAdd(&counts[i1], 1);
        tokidx [i1 * NTOK + s1] = token;
        tokgate[i1 * NTOK + s1] = p[i1] * inv;
    }
}

// ---------------------------------------------------------------------------
// WMMA fragment loaders (16-bit matrices; A and B both use contiguous k = 16*(L/16) + element)
// ---------------------------------------------------------------------------

// A-matrix 16x32 bf16 from LDS (row-major, rowstride elements).
// Lane L: row m = L%16. VGPR v holds K pair: k0 = 2v + (L>=16?16:0)
__device__ __forceinline__ v16bf load_a_frag(const __bf16* lds, int rowstride,
                                             int kofs, int lane) {
    const int m  = lane & 15;
    const int hi = (lane >> 4) & 1;
    union { v16bf v; unsigned int u[8]; } f;
    const unsigned int* row =
        (const unsigned int*)(lds + m * rowstride + kofs);  // kofs even -> dword aligned
#pragma unroll
    for (int v2 = 0; v2 < 8; ++v2) {
        const int k0 = 2 * v2 + (hi ? 16 : 0);
        f.u[v2] = row[k0 >> 1];
    }
    return f.v;
}

// B-matrix 32x16 bf16 built from fp32 global memory at (k0,n0); ld = row stride
// in floats (K runs over rows). Lane L: col n = L%16, rows kbase = (L>=16)*16,
// element j <-> k = kbase + j. Lanes 0..15 read 16 consecutive floats -> each
// dword load touches two fully-consumed 64B segments.
__device__ __forceinline__ v16bf load_b_frag_f32(const float* __restrict__ g,
                                                 size_t ld, int lane) {
    const int n  = lane & 15;
    const int kb = lane & 16;               // 0 or 16
    union { v16bf v; __bf16 h[16]; } f;
    const float* p = g + (size_t)kb * ld + n;
#pragma unroll
    for (int j = 0; j < 16; ++j) f.h[j] = (__bf16)p[(size_t)j * ld];
    return f.v;
}

// ---------------------------------------------------------------------------
// Kernel 2: fused FFN per (expert, 64-token tile).
//   h  = relu(x_tile @ w1[e] + b1[e])   (streamed through LDS in HC chunks)
//   o += h @ w2[e];  out += g * (o + b2[e])  (atomic fp32 accumulate)
// Phase 1: wave w computes h[:, w*16 .. w*16+16) of the chunk (1 B-frag -> 4 WMMAs)
// Phase 2: wave w owns D columns [w*64, w*64+64)            (1 B-frag -> 4 WMMAs)
// ---------------------------------------------------------------------------
__launch_bounds__(BLOCK, 1)
__global__ void moe_ffn_fused(const float* __restrict__ x,
                              const float* __restrict__ w1,
                              const float* __restrict__ b1,
                              const float* __restrict__ w2,
                              const float* __restrict__ b2,
                              const int*   __restrict__ counts,
                              const int*   __restrict__ tokidx,
                              const float* __restrict__ tokgate,
                              float* __restrict__ out) {
    __shared__ __bf16 x_lds[MT * D_];        // 128 KB, bf16 x tile (64 rows)
    __shared__ __bf16 h_lds[2][MT * HC];     // 2 x 32 KB, double-buffered h chunk
    __shared__ int    tok_s[MT];
    __shared__ float  g_s[MT];

    const int e    = blockIdx.y;
    const int cnt  = counts[e];
    const int base = blockIdx.x * MT;
    if (base >= cnt) return;

    const int tid  = threadIdx.x;
    const int lane = tid & 31;
    const int wave = tid >> 5;

    if (tid < MT) {
        const int idx  = base + tid;
        const bool ok  = idx < cnt;
        const int slot = ok ? idx : base;
        tok_s[tid] = tokidx[e * NTOK + slot];
        g_s[tid]   = ok ? tokgate[e * NTOK + slot] : 0.0f;   // padded rows: weight 0
    }
    __syncthreads();

    // gather x tile -> bf16 LDS (coalesced float4 per row segment)
#pragma unroll
    for (int it = 0; it < MT * (D_ / 4) / BLOCK; ++it) {     // 32 iterations
        const int u  = tid + it * BLOCK;   // float4 unit, 256 per row
        const int r  = u >> 8;
        const int c4 = u & 255;
        const float4 v = ((const float4*)(x + (size_t)tok_s[r] * D_))[c4];
        const int o = r * D_ + c4 * 4;
        x_lds[o + 0] = (__bf16)v.x;  x_lds[o + 1] = (__bf16)v.y;
        x_lds[o + 2] = (__bf16)v.z;  x_lds[o + 3] = (__bf16)v.w;
    }
    __syncthreads();

    v8f oacc[4][4];   // [mtile][ntile]
#pragma unroll
    for (int mt = 0; mt < 4; ++mt)
#pragma unroll
        for (int nt = 0; nt < 4; ++nt)
            oacc[mt][nt] = (v8f){0.f,0.f,0.f,0.f,0.f,0.f,0.f,0.f};

    const float* w1e = w1 + (size_t)e * D_ * H_;
    const float* w2e = w2 + (size_t)e * H_ * D_;
    const int ncol0 = wave * NCOL_W;   // this wave's D-column range (phase 2)
    const int hcol0 = wave * 16;       // this wave's H-column slice (phase 1)
    const int n  = lane & 15;
    const int hi = (lane >> 4) & 1;

    for (int ch = 0; ch < H_ / HC; ++ch) {
        const int hbase = ch * HC;
        __bf16* hbuf = h_lds[ch & 1];

        // ---- phase 1: h[:, hbase+hcol0 .. +16) = relu(x @ w1 + b1) ----------
        {
            v8f c[4];
#pragma unroll
            for (int mt = 0; mt < 4; ++mt)
                c[mt] = (v8f){0.f,0.f,0.f,0.f,0.f,0.f,0.f,0.f};

            const int hcol = hbase + hcol0;
            for (int k = 0; k < D_; k += 32) {
                v16bf b = load_b_frag_f32(w1e + (size_t)k * H_ + hcol, H_, lane);
#pragma unroll
                for (int mt = 0; mt < 4; ++mt) {
                    v16bf a = load_a_frag(x_lds + mt * 16 * D_, D_, k, lane);
                    c[mt] = __builtin_amdgcn_wmma_f32_16x16x32_bf16(
                                false, a, false, b, (short)0, c[mt], false, false);
                }
            }
            const float bias = b1[(size_t)e * H_ + hcol + n];
#pragma unroll
            for (int mt = 0; mt < 4; ++mt) {
#pragma unroll
                for (int v = 0; v < 8; ++v) {
                    const int m = mt * 16 + v + hi * 8;
                    float hv = c[mt][v] + bias;
                    hv = hv > 0.0f ? hv : 0.0f;
                    hbuf[m * HC + hcol0 + n] = (__bf16)hv;
                }
            }
        }
        __syncthreads();   // h chunk visible; prior chunk's reads retired

        // ---- phase 2: oacc += h_chunk @ w2[hbase:hbase+HC, ncol0:ncol0+64] --
#pragma unroll
        for (int kt = 0; kt < HC / 32; ++kt) {
            v16bf a[4];
#pragma unroll
            for (int mt = 0; mt < 4; ++mt)
                a[mt] = load_a_frag(hbuf + mt * 16 * HC, HC, kt * 32, lane);
#pragma unroll
            for (int ntile = 0; ntile < 4; ++ntile) {
                v16bf b = load_b_frag_f32(
                    w2e + (size_t)(hbase + kt * 32) * D_ + ncol0 + ntile * 16,
                    D_, lane);
#pragma unroll
                for (int mt = 0; mt < 4; ++mt) {
                    oacc[mt][ntile] = __builtin_amdgcn_wmma_f32_16x16x32_bf16(
                            false, a[mt], false, b, (short)0, oacc[mt][ntile],
                            false, false);
                }
            }
        }
    }

    // ---- epilogue: out[tok, d] += g * (oacc + b2) --------------------------
#pragma unroll
    for (int ntile = 0; ntile < 4; ++ntile) {
        const int dcol = ncol0 + ntile * 16 + n;
        const float bias = b2[(size_t)e * D_ + dcol];
#pragma unroll
        for (int mt = 0; mt < 4; ++mt) {
#pragma unroll
            for (int v = 0; v < 8; ++v) {
                const int m = mt * 16 + v + hi * 8;
                const float g = g_s[m];
                if (g != 0.0f) {
                    atomicAdd(&out[(size_t)tok_s[m] * D_ + dcol],
                              g * (oacc[mt][ntile][v] + bias));
                }
            }
        }
    }
}

// ---------------------------------------------------------------------------
// Launch
// ---------------------------------------------------------------------------
extern "C" void kernel_launch(void* const* d_in, const int* in_sizes, int n_in,
                              void* d_out, int out_size, void* d_ws, size_t ws_size,
                              hipStream_t stream) {
    const float* x      = (const float*)d_in[0];
    const float* gate_w = (const float*)d_in[1];
    const float* gate_b = (const float*)d_in[2];
    const float* w1     = (const float*)d_in[3];
    const float* b1     = (const float*)d_in[4];
    const float* w2     = (const float*)d_in[5];
    const float* b2     = (const float*)d_in[6];
    float* out = (float*)d_out;

    // workspace layout
    char* ws = (char*)d_ws;
    int*   counts  = (int*)ws;                          // 8 ints (padded to 32B)
    int*   tokidx  = (int*)(ws + 32);                   // E * NTOK ints
    float* tokgate = (float*)(ws + 32 + sizeof(int) * E_ * NTOK);

    // 0) zero output + counters
    {
        const int n = NTOK * D_;
        moe_zero_kernel<<<(n + 255) / 256, 256, 0, stream>>>(out, n, counts);
    }
    // 1) gate + top-2 routing (8 waves per block -> 8 tokens per block)
    {
        moe_gate_topk<<<NTOK / 8, 256, 0, stream>>>(
            x, gate_w, gate_b, counts, tokidx, tokgate);
    }
    // 2) fused expert FFN; worst case all 4096 tokens on one expert -> 64 tiles
    {
        dim3 grid(NTOK / MT, E_);
        moe_ffn_fused<<<grid, BLOCK, 0, stream>>>(
            x, w1, b1, w2, b2, counts, tokidx, tokgate, out);
    }
}
